// Maxout_17781164606116
// MI455X (gfx1250) — compile-verified, matrix operand order checked by reading, not run on hardware
//
#include <hip/hip_runtime.h>
#include <hip/hip_bf16.h>

// ---------------------------------------------------------------------------
// GRU (E=H=1024) over [N=64, L=512] + masked max over time -> [64, 1024] f32.
// Strategy: single persistent kernel; 256 waves each own a 16x16 tile of h.
// Per step: 6 bf16 WMMA accumulations per K-slice (x/h projections for r,z and
// separate i_n / h_n), gate math in f32, running masked max in registers,
// software grid barrier between steps. Weights pre-converted to bf16 (L2
// resident, 12 MB). A-panels staged in LDS per workgroup per step.
// ---------------------------------------------------------------------------

typedef __attribute__((ext_vector_type(16))) __bf16 v16bf;
typedef __attribute__((ext_vector_type(8)))  __bf16 v8bf;
typedef __attribute__((ext_vector_type(8)))  float  v8f;
typedef __attribute__((ext_vector_type(4)))  float  v4f;

constexpr int kE = 1024;   // embedding features
constexpr int kH = 1024;   // hidden features
constexpr int kN = 64;     // batch
constexpr int kL = 512;    // sequence length
constexpr int kGH = 3 * kH;

constexpr int kNumWG = 32;      // 32 WGs x 8 waves = 256 waves = 256 h-tiles
constexpr int kThreads = 256;

// ---------------- workspace layout ----------------
// [0, 256)                       : grid-barrier counter (unsigned)
// [256, 256 + 128KiB)            : h state, bf16, [64 x 1024]
// then Wih bf16 [3072 x 1024], then Whh bf16 [3072 x 1024]
constexpr size_t kOffH   = 256;
constexpr size_t kOffWih = kOffH + (size_t)kN * kH * sizeof(__bf16);
constexpr size_t kOffWhh = kOffWih + (size_t)kGH * kE * sizeof(__bf16);

// ---------------------------------------------------------------------------
__global__ void init_ws_kernel(__bf16* __restrict__ h_bf, unsigned* __restrict__ ctr) {
  int i = blockIdx.x * blockDim.x + threadIdx.x;
  if (i == 0) *ctr = 0u;
  for (; i < kN * kH; i += gridDim.x * blockDim.x) h_bf[i] = (__bf16)0.0f;
}

__global__ void convert_weights_kernel(const float* __restrict__ Wih,
                                       const float* __restrict__ Whh,
                                       __bf16* __restrict__ Wih_b,
                                       __bf16* __restrict__ Whh_b) {
  int i = blockIdx.x * blockDim.x + threadIdx.x;
  const int total = kGH * kE;
  for (; i < total; i += gridDim.x * blockDim.x) {
    Wih_b[i] = (__bf16)Wih[i];
    Whh_b[i] = (__bf16)Whh[i];
  }
}

// Load one 32x16 (KxN) bf16 B fragment: lane -> column n; lanes 0-15 hold
// K {0..7, 16..23}, lanes 16-31 hold K {8..15, 24..31} (16-bit WMMA layout).
__device__ __forceinline__ v16bf load_b_frag(const __bf16* __restrict__ W,
                                             int grow, int k0, int kb) {
  const __bf16* base = W + (size_t)grow * 1024 + k0 + kb;
  v8bf b0 = *(const v8bf*)(base);
  v8bf b1 = *(const v8bf*)(base + 16);
  v16bf b;
#pragma unroll
  for (int i = 0; i < 8; ++i) { b[i] = b0[i]; b[8 + i] = b1[i]; }
  return b;
}

__device__ __forceinline__ float sigmoid_f(float x) {
  return 1.0f / (1.0f + __expf(-x));
}
__device__ __forceinline__ float tanh_f(float x) {
  return 2.0f / (1.0f + __expf(-2.0f * x)) - 1.0f;
}

__launch_bounds__(kThreads, 1)
__global__ void gru_maxpool_kernel(const float* __restrict__ q_emb,
                                   const int* __restrict__ q_len,
                                   const float* __restrict__ b_ih,
                                   const float* __restrict__ b_hh,
                                   const __bf16* __restrict__ Wih,
                                   const __bf16* __restrict__ Whh,
                                   __bf16* __restrict__ h_bf,
                                   unsigned* __restrict__ ctr,
                                   float* __restrict__ out) {
  __shared__ __bf16 lds_x[16 * kE];   // x_t panel (rows of this row-block), bf16
  __shared__ __bf16 lds_h[16 * kH];   // h panel, bf16

  const int tid  = threadIdx.x;
  const int lane = tid & 31;
  const int wv   = tid >> 5;                         // wave in WG: 0..7
  const int r    = blockIdx.x >> 3;                  // row block: 0..3 (16 batch rows)
  const int c    = (blockIdx.x & 7) * 8 + wv;        // col block: 0..63 (16 h cols)
  const int j0   = c * 16;
  const int n    = lane & 15;                        // B column / C column
  const int hi   = lane >> 4;
  const int kb   = hi * 8;                           // per-lane K base within 32-slice
  const int m    = n;                                // A row within tile

  // Per-column biases (b_ih + b_hh fused where the gates sum them anyway).
  const float bias_r  = b_ih[j0 + n]            + b_hh[j0 + n];
  const float bias_z  = b_ih[kH + j0 + n]       + b_hh[kH + j0 + n];
  const float bias_in = b_ih[2 * kH + j0 + n];
  const float bias_hn = b_hh[2 * kH + j0 + n];

  // Sequence lengths for the 8 batch rows this lane's accumulator covers.
  int qv[8];
#pragma unroll
  for (int v = 0; v < 8; ++v) qv[v] = q_len[r * 16 + v + 8 * hi];

  v8f hprev, hmax;
#pragma unroll
  for (int v = 0; v < 8; ++v) { hprev[v] = 0.0f; hmax[v] = -3.402823466e38f; }

  for (int t = 0; t < kL; ++t) {
    // ---- stage A panels (16 rows x 1024) into LDS: x_t (f32->bf16) and h ----
    for (int q = tid; q < 2048; q += kThreads) {
      const int row = q >> 7;            // 0..15
      const int k   = (q & 127) * 8;     // 0..1016 step 8
      const float* src = q_emb + ((size_t)(r * 16 + row) * kL + t) * kE + k;
      v4f f0 = *(const v4f*)(src);
      v4f f1 = *(const v4f*)(src + 4);
      v8bf xb;
#pragma unroll
      for (int i = 0; i < 4; ++i) { xb[i] = (__bf16)f0[i]; xb[4 + i] = (__bf16)f1[i]; }
      *(v8bf*)&lds_x[row * kE + k] = xb;
      *(v8bf*)&lds_h[row * kH + k] = *(const v8bf*)&h_bf[(size_t)(r * 16 + row) * kH + k];
    }
    __syncthreads();

    // ---- 6 fused GEMM tiles over K=1024 ----
    v8f acc_r, acc_z, acc_in, acc_hn;
#pragma unroll
    for (int v = 0; v < 8; ++v) {
      acc_r[v] = bias_r; acc_z[v] = bias_z; acc_in[v] = bias_in; acc_hn[v] = bias_hn;
    }

    for (int k0 = 0; k0 < kE; k0 += 32) {
      // A fragments (16x32) from LDS, per documented 16-bit layout.
      v8bf x0 = *(const v8bf*)&lds_x[m * kE + k0 + kb];
      v8bf x1 = *(const v8bf*)&lds_x[m * kE + k0 + kb + 16];
      v8bf h0 = *(const v8bf*)&lds_h[m * kH + k0 + kb];
      v8bf h1 = *(const v8bf*)&lds_h[m * kH + k0 + kb + 16];
      v16bf ax, ah;
#pragma unroll
      for (int i = 0; i < 8; ++i) {
        ax[i] = x0[i]; ax[8 + i] = x1[i];
        ah[i] = h0[i]; ah[8 + i] = h1[i];
      }

      v16bf bxr = load_b_frag(Wih, j0 + n,            k0, kb);
      v16bf bxz = load_b_frag(Wih, kH + j0 + n,       k0, kb);
      v16bf bxn = load_b_frag(Wih, 2 * kH + j0 + n,   k0, kb);
      v16bf bhr = load_b_frag(Whh, j0 + n,            k0, kb);
      v16bf bhz = load_b_frag(Whh, kH + j0 + n,       k0, kb);
      v16bf bhn = load_b_frag(Whh, 2 * kH + j0 + n,   k0, kb);

      acc_r  = __builtin_amdgcn_wmma_f32_16x16x32_bf16(false, ax, false, bxr, (short)0, acc_r,  false, false);
      acc_r  = __builtin_amdgcn_wmma_f32_16x16x32_bf16(false, ah, false, bhr, (short)0, acc_r,  false, false);
      acc_z  = __builtin_amdgcn_wmma_f32_16x16x32_bf16(false, ax, false, bxz, (short)0, acc_z,  false, false);
      acc_z  = __builtin_amdgcn_wmma_f32_16x16x32_bf16(false, ah, false, bhz, (short)0, acc_z,  false, false);
      acc_in = __builtin_amdgcn_wmma_f32_16x16x32_bf16(false, ax, false, bxn, (short)0, acc_in, false, false);
      acc_hn = __builtin_amdgcn_wmma_f32_16x16x32_bf16(false, ah, false, bhn, (short)0, acc_hn, false, false);
    }

    // ---- gate math, masked running max, publish h_new ----
    v8f hnew;
#pragma unroll
    for (int v = 0; v < 8; ++v) {
      float rg  = sigmoid_f(acc_r[v]);
      float zg  = sigmoid_f(acc_z[v]);
      float ng  = tanh_f(acc_in[v] + rg * acc_hn[v]);
      float hv  = (1.0f - zg) * ng + zg * hprev[v];
      hnew[v] = hv;
      if (t < qv[v]) hmax[v] = fmaxf(hmax[v], hv);
    }
    hprev = hnew;
#pragma unroll
    for (int v = 0; v < 8; ++v) {
      const int row = r * 16 + v + 8 * hi;
      h_bf[(size_t)row * kH + j0 + n] = (__bf16)hnew[v];
    }

    // ---- grid barrier (32 co-resident WGs) ----
    __threadfence();
    __syncthreads();
    if (tid == 0) {
      __hip_atomic_fetch_add(ctr, 1u, __ATOMIC_ACQ_REL, __HIP_MEMORY_SCOPE_AGENT);
      const unsigned target = (unsigned)(t + 1) * (unsigned)gridDim.x;
      while (__hip_atomic_load(ctr, __ATOMIC_ACQUIRE, __HIP_MEMORY_SCOPE_AGENT) < target)
        __builtin_amdgcn_s_sleep(1);
    }
    __syncthreads();
    __threadfence();
  }

  // ---- write masked-max output tile ----
#pragma unroll
  for (int v = 0; v < 8; ++v) {
    const int row = r * 16 + v + 8 * hi;
    out[(size_t)row * kH + j0 + n] = hmax[v];
  }
}

// ---------------------------------------------------------------------------
extern "C" void kernel_launch(void* const* d_in, const int* in_sizes, int n_in,
                              void* d_out, int out_size, void* d_ws, size_t ws_size,
                              hipStream_t stream) {
  (void)in_sizes; (void)n_in; (void)out_size; (void)ws_size;

  const float* q_emb = (const float*)d_in[0];   // [64, 512, 1024] f32
  const int*   q_len = (const int*)d_in[1];     // [64] i32
  const float* W_ih  = (const float*)d_in[2];   // [3072, 1024] f32
  const float* W_hh  = (const float*)d_in[3];   // [3072, 1024] f32
  const float* b_ih  = (const float*)d_in[4];   // [3072] f32
  const float* b_hh  = (const float*)d_in[5];   // [3072] f32
  float*       out   = (float*)d_out;           // [64, 1024] f32

  char* ws = (char*)d_ws;
  unsigned* ctr   = (unsigned*)ws;
  __bf16*   h_bf  = (__bf16*)(ws + kOffH);
  __bf16*   Wih_b = (__bf16*)(ws + kOffWih);
  __bf16*   Whh_b = (__bf16*)(ws + kOffWhh);

  init_ws_kernel<<<64, 256, 0, stream>>>(h_bf, ctr);
  convert_weights_kernel<<<1024, 256, 0, stream>>>(W_ih, W_hh, Wih_b, Whh_b);
  gru_maxpool_kernel<<<kNumWG, kThreads, 0, stream>>>(
      q_emb, q_len, b_ih, b_hh, Wih_b, Whh_b, h_bf, ctr, out);
}
